// Decoder_8495445311803
// MI455X (gfx1250) — compile-verified, matrix operand order checked by reading, not run on hardware
//
#include <hip/hip_runtime.h>

typedef _Float16 f16;
typedef __attribute__((ext_vector_type(16))) _Float16 v16h;
typedef __attribute__((ext_vector_type(8)))  _Float16 v8h;
typedef __attribute__((ext_vector_type(8)))  float    v8f;

// Problem constants (from reference)
#define NB   4096
#define VV   26
#define CC   256
#define K3   3
#define TSN  25
#define MROWS (NB*VV)          // 106496
#define KAUG 288               // 256 m-channels + 9 x + pad -> 9 K-tiles of 32
#define GOUT 1024              // 4 gates * 256 channels, interleaved ch*4+gate
#define MB   32                // rows per workgroup in GRU kernel

// ---------------- WMMA helpers (gfx1250, wave32) ----------------

__device__ __forceinline__ v8f wmma16(v16h a, v16h b, v8f c) {
  // D = A(16x32 f16) x B(32x16 f16) + C(16x16 f32)
  return __builtin_amdgcn_wmma_f32_16x16x32_f16(false, a, false, b, (short)0, c, false, false);
}

// A fragment: rows striped per lane. lane<16: row=lane, K = {g*8..g*8+7, 16+g*8..}
__device__ __forceinline__ v16h ld_a16(const f16* base, int ldk) {
  int lane = threadIdx.x & 31;
  int g  = (lane >> 4) << 3;   // 0 or 8
  int rr = lane & 15;
  const f16* p = base + rr * ldk + g;
  v8h lo = *(const v8h*)(p);
  v8h hi = *(const v8h*)(p + 16);
  v16h o;
#pragma unroll
  for (int i = 0; i < 8; ++i) { o[i] = lo[i]; o[i + 8] = hi[i]; }
  return o;
}

// B fragment from weight matrix W[out][in] (row-major): B[k][n] = W[n][k]
__device__ __forceinline__ v16h ld_b16(const f16* w, int colBase, int ldk, int k0) {
  int lane = threadIdx.x & 31;
  int g  = (lane >> 4) << 3;
  int cc = lane & 15;
  const f16* p = w + (size_t)(colBase + cc) * ldk + k0 + g;
  v8h lo = *(const v8h*)(p);
  v8h hi = *(const v8h*)(p + 16);
  v16h o;
#pragma unroll
  for (int i = 0; i < 8; ++i) { o[i] = lo[i]; o[i + 8] = hi[i]; }
  return o;
}

__device__ __forceinline__ float nz(int s, int r, int j) {
  unsigned h = (unsigned)(s * 0x9E3779B1u) ^ (unsigned)(r * 0x85EBCA77u) ^ (unsigned)(j * 0xC2B2AE3Du);
  h ^= h >> 16; h *= 0x7FEB352Du; h ^= h >> 15; h *= 0x846CA68Bu; h ^= h >> 16;
  float u1 = (float)(h & 0xFFFFFFu) * (1.0f / 16777216.0f) - 0.5f;
  unsigned h2 = h * 0x9E3779B1u; h2 ^= h2 >> 16;
  float u2 = (float)(h2 & 0xFFFFFFu) * (1.0f / 16777216.0f) - 0.5f;
  return (u1 + u2) * 1.22e-3f;   // ~ N(0, 5e-4)
}

// ---------------- prep kernels ----------------

__global__ __launch_bounds__(256) void prep_small_kernel(
    const float* __restrict__ A, const float* __restrict__ emul,
    const float* __restrict__ eadd, const float* __restrict__ conv_b,
    const float* __restrict__ W3,
    float* __restrict__ A_adj, float* __restrict__ bmT, f16* __restrict__ W3h) {
  __shared__ float As[K3 * VV * VV];
  __shared__ float cs[K3 * VV];
  int tid = threadIdx.x;
  for (int i = tid; i < K3 * VV * VV; i += 256) As[i] = A[i] * emul[i] + eadd[i];
  __syncthreads();
  for (int i = tid; i < K3 * VV * VV; i += 256) A_adj[i] = As[i];
  for (int i = tid; i < K3 * VV; i += 256) {
    int k = i / VV, v = i % VV;
    float s = 0.f;
    for (int w = 0; w < VV; ++w) s += As[k * VV * VV + w * VV + v];
    cs[i] = s;
  }
  __syncthreads();
  for (int i = tid; i < VV * CC; i += 256) {
    int v = i >> 8, c = i & 255;
    float s = 0.f;
    for (int k = 0; k < K3; ++k) s += conv_b[k * CC + c] * cs[k * VV + v];
    bmT[i] = s;   // bias term broadcast through A, added in gcn epilogue
  }
  for (int i = tid; i < 16 * CC; i += 256) {
    int row = i >> 8, c = i & 255;
    W3h[i] = (f16)(row < 3 ? W3[row * CC + c] : 0.f);
  }
}

__global__ __launch_bounds__(256) void prep_weights_kernel(
    const float* __restrict__ Wir, const float* __restrict__ Wii, const float* __restrict__ Win,
    const float* __restrict__ Whr, const float* __restrict__ Whi, const float* __restrict__ Whh,
    const float* __restrict__ conv_w, const float* __restrict__ W1, const float* __restrict__ W2,
    f16* __restrict__ Wg, f16* __restrict__ Gw, f16* __restrict__ W1h, f16* __restrict__ W2h) {
  const int nWg = GOUT * KAUG;          // 294912
  const int nGw = CC * (K3 * CC);       // 196608
  const int nW  = CC * CC;              // 65536
  const int total = nWg + nGw + 2 * nW;
  for (int i = blockIdx.x * blockDim.x + threadIdx.x; i < total; i += gridDim.x * blockDim.x) {
    if (i < nWg) {
      int out = i / KAUG, kin = i % KAUG;
      int ch = out >> 2, gate = out & 3;
      float w = 0.f;
      if (kin < CC) {               // m-channel part
        if      (gate == 0) w = Whr[ch * CC + kin];
        else if (gate == 1) w = Whi[ch * CC + kin];
        else if (gate == 3) w = Whh[ch * CC + kin];
      } else if (kin < CC + 9) {    // x part
        int j = kin - CC;
        if      (gate == 0) w = Wir[ch * 9 + j];
        else if (gate == 1) w = Wii[ch * 9 + j];
        else if (gate == 2) w = Win[ch * 9 + j];
      }
      Wg[i] = (f16)w;
    } else if (i < nWg + nGw) {
      int j = i - nWg;
      int c = j / (K3 * CC); int rem = j % (K3 * CC);
      int k = rem >> 8; int cp = rem & 255;
      Gw[j] = (f16)conv_w[(k * CC + c) * CC + cp];   // Gw[c][k*256+c'] = conv_w[kC+c][c']
    } else if (i < nWg + nGw + nW) {
      int j = i - nWg - nGw; W1h[j] = (f16)W1[j];
    } else {
      int j = i - nWg - nGw - nW; W2h[j] = (f16)W2[j];
    }
  }
}

__global__ __launch_bounds__(256) void prep_state_kernel(
    const float* __restrict__ hidden, const float* __restrict__ x0,
    const float* __restrict__ xp, const float* __restrict__ xp2,
    float* __restrict__ hidS, float* __restrict__ hb0,
    float* __restrict__ hb1, float* __restrict__ hb2) {
  const int totalH = NB * VV * CC;
  int stride = gridDim.x * blockDim.x;
  for (int i = blockIdx.x * blockDim.x + threadIdx.x; i < totalH; i += stride) {
    int n = i / (VV * CC); int rem = i % (VV * CC);
    int v = rem / CC;      int c = rem % CC;
    hidS[i] = hidden[(size_t)n * CC * VV + c * VV + v];   // [N,C,V] -> [N,V,C]
  }
  for (int i = blockIdx.x * blockDim.x + threadIdx.x; i < MROWS * 3; i += stride) {
    hb0[i] = x0[i]; hb1[i] = xp[i]; hb2[i] = xp2[i];
  }
}

// ---------------- gcn: msg16[N,V,C] = gcn(hidS), f16 output ----------------

__global__ __launch_bounds__(256) void gcn_kernel(
    const float* __restrict__ hidS, const float* __restrict__ A_adj,
    const f16* __restrict__ Gw, const float* __restrict__ bmT,
    f16* __restrict__ msg16) {
  __shared__ __align__(16) float As[K3 * VV * VV];   // 8112 B
  __shared__ __align__(16) f16 xg[32 * (K3 * CC)];   // 32x768 f16 = 49152 B
  const int tid = threadIdx.x;
  const int wave = tid >> 5;
  const int lane = tid & 31;
  const int n = blockIdx.x;

  for (int i = tid; i < K3 * VV * VV; i += 256) As[i] = A_adj[i];
  for (int i = tid; i < 6 * (K3 * CC); i += 256) xg[VV * (K3 * CC) + i] = (f16)0.f; // pad rows 26..31
  __syncthreads();

  const float* hidn = hidS + (size_t)n * VV * CC;
  // xg[v][k*256+c] = sum_w A_adj[k][w][v] * hid[n][w][c]
  for (int o = tid; o < VV * (K3 * CC); o += 256) {
    int v = o / (K3 * CC); int rem = o % (K3 * CC);
    int k = rem >> 8;      int c = rem & 255;
    float s = 0.f;
    for (int w = 0; w < VV; ++w) s += As[k * VV * VV + w * VV + v] * hidn[w * CC + c];
    xg[o] = (f16)s;
  }
  __syncthreads();

  // msgT[32x256] = xg[32x768] @ Gw^T  (WMMA, 2 M-tiles x 16 col-tiles, K=24 tiles)
  for (int t = wave; t < 32; t += 8) {
    int mi = t & 1, ct = t >> 1;
    v8f acc;
#pragma unroll
    for (int j = 0; j < 8; ++j) acc[j] = 0.f;
    for (int kb = 0; kb < 24; ++kb) {
      v16h a = ld_a16(xg + mi * 16 * (K3 * CC) + kb * 32, K3 * CC);
      v16h b = ld_b16(Gw, ct * 16, K3 * CC, kb * 32);
      acc = wmma16(a, b, acc);
    }
    int g = lane >> 4, cc = lane & 15;
    int col = ct * 16 + cc;
#pragma unroll
    for (int j = 0; j < 8; ++j) {
      int row = mi * 16 + j + 8 * g;
      if (row < VV) {
        float v = acc[j] + bmT[row * CC + col];
        msg16[((size_t)n * VV + row) * CC + col] = (f16)v;
      }
    }
  }
}

// ---------------- fused GRU + MLP step ----------------

#define SM_A   0                         // a_lds [32][288] f16 (18432B) / later hd1 [32][256] f16
#define SM_G   18432                     // per-wave 16x16 f32 scratch (8KB) / later hd2 [32][256] f16 (16384B)
#define SM_HN  (18432 + 16384)           // hn_lds [32][256] f16 (16384B)
#define SM_X   (18432 + 16384 + 16384)   // x_lds [32][4] f32 (512B)
#define SM_TOT (SM_X + 512)

__global__ __launch_bounds__(256) void gru_step_kernel(
    float* __restrict__ hidS, const f16* __restrict__ msg16, int use_msg,
    const f16* __restrict__ Wg, const f16* __restrict__ W1h,
    const f16* __restrict__ W2h, const f16* __restrict__ W3h,
    const float* __restrict__ bir, const float* __restrict__ bii,
    const float* __restrict__ b_in, const float* __restrict__ b1,
    const float* __restrict__ b2, const float* __restrict__ b3,
    const float* __restrict__ h1, const float* __restrict__ h2,
    const float* __restrict__ h3, float* __restrict__ h1next,
    float* __restrict__ out, int s) {
  __shared__ __align__(16) unsigned char smem[SM_TOT];
  f16*   a_lds  = (f16*)(smem + SM_A);
  float* g_lds  = (float*)(smem + SM_G);
  f16*   hn_lds = (f16*)(smem + SM_HN);
  float* x_lds  = (float*)(smem + SM_X);

  const int tid = threadIdx.x;
  const int wave = tid >> 5;
  const int lane = tid & 31;
  const int R0 = blockIdx.x * MB;

  // ---- stage A operand: [m(256) | x(9) | 0-pad] per row ----
  for (int i = tid; i < MB * CC; i += 256) {
    int row = i >> 8, c = i & 255;
    size_t gi = (size_t)(R0 + row) * CC + c;
    a_lds[row * KAUG + c] = use_msg ? msg16[gi] : (f16)hidS[gi];
  }
  if (tid < MB) {
    int grow = R0 + tid;
    float p[3], q[3], w[3], xv[9];
#pragma unroll
    for (int j = 0; j < 3; ++j) {
      p[j] = h1[grow * 3 + j]; q[j] = h2[grow * 3 + j]; w[j] = h3[grow * 3 + j];
    }
#pragma unroll
    for (int j = 0; j < 3; ++j) {
      xv[j]     = p[j] + nz(s, grow, j);          // ins_p + noise
      xv[3 + j] = q[j] - w[j];                    // ins_v
      xv[6 + j] = p[j] - 2.f * q[j] + w[j];       // ins_a
    }
#pragma unroll
    for (int j = 0; j < 9; ++j) a_lds[tid * KAUG + CC + j] = (f16)xv[j];
    for (int j = 9; j < 32; ++j) a_lds[tid * KAUG + CC + j] = (f16)0.f;
#pragma unroll
    for (int j = 0; j < 3; ++j) x_lds[tid * 4 + j] = xv[j];
  }
  __syncthreads();

  // ---- gate GEMM [32x288]x[288x1024] + GRU nonlinearity ----
  float* sc = g_lds + wave * 256;    // 16x16 wave-private scratch
  for (int t = wave; t < 128; t += 8) {
    int mi = t & 1, ct = t >> 1;     // ct covers channels ct*4..ct*4+3 (x4 gates)
    v8f acc;
#pragma unroll
    for (int j = 0; j < 8; ++j) acc[j] = 0.f;
    for (int kb = 0; kb < 9; ++kb) {
      v16h a = ld_a16(a_lds + mi * 16 * KAUG + kb * 32, KAUG);
      v16h b = ld_b16(Wg, ct * 16, KAUG, kb * 32);
      acc = wmma16(a, b, acc);
    }
    int g = lane >> 4, cc = lane & 15;
#pragma unroll
    for (int j = 0; j < 8; ++j) sc[(j + 8 * g) * 16 + cc] = acc[j];
#pragma unroll
    for (int p = 0; p < 2; ++p) {
      int idx = lane * 2 + p;        // 0..63 -> (row 0..15, chL 0..3)
      int row = idx & 15, chL = idx >> 4;
      int ch = ct * 4 + chL;
      float g0 = sc[row * 16 + chL * 4 + 0];   // r pre-act
      float g1 = sc[row * 16 + chL * 4 + 1];   // z pre-act
      float g2 = sc[row * 16 + chL * 4 + 2];   // x@Win^T
      float g3 = sc[row * 16 + chL * 4 + 3];   // m@Whh^T
      int lrow = mi * 16 + row;
      size_t gi = (size_t)(R0 + lrow) * CC + ch;
      float hp = hidS[gi];
      float rg = 1.f / (1.f + __expf(-(g0 + bir[ch])));
      float zg = 1.f / (1.f + __expf(-(g1 + bii[ch])));
      float nn = tanhf(g2 + b_in[ch] + rg * g3);
      float hnew = (1.f - zg) * nn + zg * hp;
      hidS[gi] = hnew;
      hn_lds[lrow * CC + ch] = (f16)hnew;
    }
  }
  __syncthreads();

  // ---- MLP layer 1: hd1 = lrelu(hnew @ W1^T + b1) ----
  f16* hd1 = a_lds;                  // reuse region 0 as [32][256] f16
  for (int t = wave; t < 32; t += 8) {
    int mi = t & 1, ct = t >> 1;
    v8f acc;
#pragma unroll
    for (int j = 0; j < 8; ++j) acc[j] = 0.f;
    for (int kb = 0; kb < 8; ++kb) {
      v16h a = ld_a16(hn_lds + mi * 16 * CC + kb * 32, CC);
      v16h b = ld_b16(W1h, ct * 16, CC, kb * 32);
      acc = wmma16(a, b, acc);
    }
    int g = lane >> 4, cc = lane & 15;
    int col = ct * 16 + cc;
    float bv = b1[col];
#pragma unroll
    for (int j = 0; j < 8; ++j) {
      float v = acc[j] + bv;
      v = v > 0.f ? v : 0.1f * v;
      hd1[(mi * 16 + j + 8 * g) * CC + col] = (f16)v;
    }
  }
  __syncthreads();

  // ---- MLP layer 2: hd2 = lrelu(hd1 @ W2^T + b2) ----
  f16* hd2 = (f16*)g_lds;            // reuse region 1 as [32][256] f16
  for (int t = wave; t < 32; t += 8) {
    int mi = t & 1, ct = t >> 1;
    v8f acc;
#pragma unroll
    for (int j = 0; j < 8; ++j) acc[j] = 0.f;
    for (int kb = 0; kb < 8; ++kb) {
      v16h a = ld_a16(hd1 + mi * 16 * CC + kb * 32, CC);
      v16h b = ld_b16(W2h, ct * 16, CC, kb * 32);
      acc = wmma16(a, b, acc);
    }
    int g = lane >> 4, cc = lane & 15;
    int col = ct * 16 + cc;
    float bv = b2[col];
#pragma unroll
    for (int j = 0; j < 8; ++j) {
      float v = acc[j] + bv;
      v = v > 0.f ? v : 0.1f * v;
      hd2[(mi * 16 + j + 8 * g) * CC + col] = (f16)v;
    }
  }
  __syncthreads();

  // ---- MLP layer 3 (padded to 16 cols) + pred write ----
  if (wave < 2) {
    int mi = wave;
    v8f acc;
#pragma unroll
    for (int j = 0; j < 8; ++j) acc[j] = 0.f;
    for (int kb = 0; kb < 8; ++kb) {
      v16h a = ld_a16(hd2 + mi * 16 * CC + kb * 32, CC);
      v16h b = ld_b16(W3h, 0, CC, kb * 32);
      acc = wmma16(a, b, acc);
    }
    int g = lane >> 4, cc = lane & 15;
    if (cc < 3) {
      float bv = b3[cc];
#pragma unroll
      for (int j = 0; j < 8; ++j) {
        int row = mi * 16 + j + 8 * g;
        float pr = x_lds[row * 4 + cc] + acc[j] + bv;   // pred = (ins_p+nz) + res
        int grow = R0 + row;
        out[((size_t)grow * TSN + s) * 3 + cc] = pr;    // [N,V,t,3]
        h1next[grow * 3 + cc] = pr;                     // rotate history
      }
    }
  }
}

// ---------------- host launcher ----------------

extern "C" void kernel_launch(void* const* d_in, const int* in_sizes, int n_in,
                              void* d_out, int out_size, void* d_ws, size_t ws_size,
                              hipStream_t stream) {
  (void)in_sizes; (void)n_in; (void)out_size; (void)ws_size;
  const float* inputs  = (const float*)d_in[0];
  const float* inp_p   = (const float*)d_in[1];
  const float* inp_p2  = (const float*)d_in[2];
  const float* hidden  = (const float*)d_in[3];
  const float* A       = (const float*)d_in[4];
  const float* emul    = (const float*)d_in[5];
  const float* eadd    = (const float*)d_in[6];
  const float* conv_w  = (const float*)d_in[7];
  const float* conv_b  = (const float*)d_in[8];
  const float* Wir     = (const float*)d_in[9];
  const float* bir     = (const float*)d_in[10];
  const float* Wii     = (const float*)d_in[11];
  const float* bii     = (const float*)d_in[12];
  const float* Win     = (const float*)d_in[13];
  const float* b_in    = (const float*)d_in[14];
  const float* Whr     = (const float*)d_in[15];
  const float* Whi     = (const float*)d_in[16];
  const float* Whh     = (const float*)d_in[17];
  const float* W1      = (const float*)d_in[18];
  const float* b1      = (const float*)d_in[19];
  const float* W2      = (const float*)d_in[20];
  const float* b2      = (const float*)d_in[21];
  const float* W3      = (const float*)d_in[22];
  const float* b3      = (const float*)d_in[23];

  char* ws = (char*)d_ws;
  size_t off = 0;
  auto carve = [&](size_t bytes) -> char* {
    char* p = ws + off;
    off += (bytes + 255) & ~(size_t)255;
    return p;
  };
  float* hidS  = (float*)carve((size_t)MROWS * CC * 4);   // [N,V,C] f32 state
  f16*   msg16 = (f16*)  carve((size_t)MROWS * CC * 2);   // [N,V,C] f16 gcn output
  float* hb[3];
  hb[0] = (float*)carve((size_t)MROWS * 3 * 4);
  hb[1] = (float*)carve((size_t)MROWS * 3 * 4);
  hb[2] = (float*)carve((size_t)MROWS * 3 * 4);
  float* A_adj = (float*)carve(K3 * VV * VV * 4);
  float* bmT   = (float*)carve(VV * CC * 4);
  f16*   Gw    = (f16*)carve((size_t)CC * K3 * CC * 2);
  f16*   Wg    = (f16*)carve((size_t)GOUT * KAUG * 2);
  f16*   W1h   = (f16*)carve((size_t)CC * CC * 2);
  f16*   W2h   = (f16*)carve((size_t)CC * CC * 2);
  f16*   W3h   = (f16*)carve((size_t)16 * CC * 2);

  prep_small_kernel<<<1, 256, 0, stream>>>(A, emul, eadd, conv_b, W3, A_adj, bmT, W3h);
  prep_weights_kernel<<<512, 256, 0, stream>>>(Wir, Wii, Win, Whr, Whi, Whh, conv_w,
                                               W1, W2, Wg, Gw, W1h, W2h);
  prep_state_kernel<<<2048, 256, 0, stream>>>(hidden, inputs, inp_p, inp_p2,
                                              hidS, hb[0], hb[1], hb[2]);

  const int p1[3] = {0, 2, 1};
  const int p2[3] = {1, 0, 2};
  const int p3[3] = {2, 1, 0};
  for (int s = 0; s < TSN; ++s) {
    if (s < 10)
      gcn_kernel<<<NB, 256, 0, stream>>>(hidS, A_adj, Gw, bmT, msg16);
    int r3 = s % 3;
    gru_step_kernel<<<MROWS / MB, 256, 0, stream>>>(
        hidS, msg16, (s < 10) ? 1 : 0, Wg, W1h, W2h, W3h,
        bir, bii, b_in, b1, b2, b3,
        hb[p1[r3]], hb[p2[r3]], hb[p3[r3]], hb[p3[r3]],
        (float*)d_out, s);
  }
}
